// FlexibleGAT_89532888252424
// MI455X (gfx1250) — matrix operand order repaired and checked by reading, not compiled, not run on hardware
//
#include <hip/hip_runtime.h>
#include <cmath>
#include <stdint.h>

typedef __attribute__((ext_vector_type(16))) __bf16 v16bf;
typedef __attribute__((ext_vector_type(8)))  __bf16 v8bf;
typedef __attribute__((ext_vector_type(8)))  float  v8f;

__device__ __forceinline__ unsigned short bfbits(float f) {
    unsigned u = __float_as_uint(f);
    return (unsigned short)((u + 0x7FFFu + ((u >> 16) & 1u)) >> 16);  // RNE
}
__device__ __forceinline__ unsigned pack2bf(float a, float b) {
    return (unsigned)bfbits(a) | ((unsigned)bfbits(b) << 16);
}

// ---------------------------------------------------------------------------
// Tiled GEMM: C[M,N] = A[M,K] @ B[K,N]  (fp32 in, bf16 WMMA, fp32 accumulate)
// Block tile 128x64, 8 waves, each wave owns a 32x32 C region (4 wmma accs).
// OOB rows/cols are address-clamped (results discarded at guarded store), so
// staging is branch-free. Next tile is prefetched into VGPRs (sw pipeline).
// ---------------------------------------------------------------------------
#define BM 128
#define BN 64
#define BK 32
#define LPAD 8   // rows padded to 40 bf16 (80B): 16B-aligned, spreads LDS banks

__global__ __launch_bounds__(256) void gemm_bf16_wmma(
    const float* __restrict__ A, const float* __restrict__ B,
    float* __restrict__ C, int M, int N, int K,
    const float* __restrict__ bias, int relu)
{
    __shared__ __attribute__((aligned(16))) __bf16 As[BM][BK + LPAD];
    __shared__ __attribute__((aligned(16))) __bf16 Bs[BN][BK + LPAD];  // transposed: [n][k]

    const int tid  = threadIdx.x;
    const int lane = tid & 31;
    const int wid  = tid >> 5;      // 0..7
    const int wm   = wid & 3;       // M subtile (4)
    const int wn   = wid >> 2;      // N subtile (2)
    const int m0   = blockIdx.y * BM;
    const int n0   = blockIdx.x * BN;

    // ---- staging coordinates (branch-free: clamp addresses) ----
    // A: 4 x float4 per thread. idx -> row = idx/8, c4 = idx%8 (8 float4 per 32-f row)
    const float* aPtr[4];
    int aRow[4], aC4[4];
    #pragma unroll
    for (int i = 0; i < 4; ++i) {
        int idx = tid + i * 256;
        int r = idx >> 3, c4 = idx & 7;
        int gr = m0 + r; if (gr > M - 1) gr = M - 1;          // clamp: row discarded later
        aPtr[i] = A + (size_t)gr * K + c4 * 4;
        aRow[i] = r; aC4[i] = c4;
    }
    // B: 2 groups of 4 k-consecutive elements. idx -> n = idx%64, k4 = idx/64
    const float* bPtr[2];
    int bN[2], bK4[2];
    #pragma unroll
    for (int i = 0; i < 2; ++i) {
        int idx = tid + i * 256;
        int n = idx & 63, k4 = idx >> 6;
        int gn = n0 + n; if (gn > N - 1) gn = N - 1;          // clamp: col discarded later
        bPtr[i] = B + (size_t)(k4 * 4) * N + gn;
        bN[i] = n; bK4[i] = k4;
    }

    v8f acc[2][2] = {};
    float4 aReg[4];
    float  bReg[2][4];

    // prefetch first tile
    #pragma unroll
    for (int i = 0; i < 4; ++i) aReg[i] = *reinterpret_cast<const float4*>(aPtr[i]);
    #pragma unroll
    for (int i = 0; i < 2; ++i)
        #pragma unroll
        for (int j = 0; j < 4; ++j) bReg[i][j] = bPtr[i][(size_t)j * N];

    for (int k0 = 0; k0 < K; k0 += BK) {
        // ---- commit prefetched tile to LDS (packed bf16, ds_store_b64) ----
        #pragma unroll
        for (int i = 0; i < 4; ++i) {
            uint2 u; u.x = pack2bf(aReg[i].x, aReg[i].y); u.y = pack2bf(aReg[i].z, aReg[i].w);
            *reinterpret_cast<uint2*>(&As[aRow[i]][aC4[i] * 4]) = u;
        }
        #pragma unroll
        for (int i = 0; i < 2; ++i) {
            uint2 u; u.x = pack2bf(bReg[i][0], bReg[i][1]); u.y = pack2bf(bReg[i][2], bReg[i][3]);
            *reinterpret_cast<uint2*>(&Bs[bN[i]][bK4[i] * 4]) = u;
        }
        __syncthreads();

        // ---- prefetch next tile (overlaps with ds_load + wmma below) ----
        if (k0 + BK < K) {
            #pragma unroll
            for (int i = 0; i < 4; ++i) {
                aPtr[i] += BK;
                aReg[i] = *reinterpret_cast<const float4*>(aPtr[i]);
            }
            #pragma unroll
            for (int i = 0; i < 2; ++i) {
                bPtr[i] += (size_t)BK * N;
                #pragma unroll
                for (int j = 0; j < 4; ++j) bReg[i][j] = bPtr[i][(size_t)j * N];
            }
        }

        // ---- fragments (aligned ds_load_b128) + 4x wmma ----
        const int l15 = lane & 15;
        const int akb = (lane >> 4) * 8;    // A: lo-lanes k=0..7/16..23, hi-lanes k=8../24..
        const int bkb = (lane >> 4) * 8;    // B: lo-lanes k=0..7/16..23, hi-lanes k=8../24..

        v16bf afr[2], bfr[2];
        #pragma unroll
        for (int mi = 0; mi < 2; ++mi) {
            const __bf16* ap = &As[wm * 32 + mi * 16 + l15][0];
            v8bf lo = *reinterpret_cast<const v8bf*>(ap + akb);
            v8bf hi = *reinterpret_cast<const v8bf*>(ap + akb + 16);
            #pragma unroll
            for (int j = 0; j < 8; ++j) { afr[mi][j] = lo[j]; afr[mi][j + 8] = hi[j]; }
        }
        #pragma unroll
        for (int ni = 0; ni < 2; ++ni) {
            const __bf16* bp = &Bs[wn * 32 + ni * 16 + l15][0];
            v8bf lo = *reinterpret_cast<const v8bf*>(bp + bkb);
            v8bf hi = *reinterpret_cast<const v8bf*>(bp + bkb + 16);
            #pragma unroll
            for (int j = 0; j < 8; ++j) { bfr[ni][j] = lo[j]; bfr[ni][j + 8] = hi[j]; }
        }
        #pragma unroll
        for (int mi = 0; mi < 2; ++mi)
            #pragma unroll
            for (int ni = 0; ni < 2; ++ni)
                acc[mi][ni] = __builtin_amdgcn_wmma_f32_16x16x32_bf16(
                    false, afr[mi], false, bfr[ni], (short)0, acc[mi][ni], false, false);
        __syncthreads();
    }

    // ---- epilogue: lane = col(lane&15), acc element r = row (+8 for hi half) ----
    const int lc = lane & 15;
    const int lr = (lane >> 4) * 8;
    #pragma unroll
    for (int mi = 0; mi < 2; ++mi)
        #pragma unroll
        for (int ni = 0; ni < 2; ++ni) {
            int gcol = n0 + wn * 32 + ni * 16 + lc;
            if (gcol >= N) continue;
            float bv = bias ? bias[gcol] : 0.f;
            #pragma unroll
            for (int r = 0; r < 8; ++r) {
                int grow = m0 + wm * 32 + mi * 16 + lr + r;
                if (grow < M) {
                    float v = acc[mi][ni][r] + bv;
                    if (relu) v = fmaxf(v, 0.f);
                    C[(size_t)grow * N + gcol] = v;
                }
            }
        }
}

// ---------------------------------------------------------------------------
__global__ void fill_kernel(float* p, float v, int n) {
    int i = blockIdx.x * blockDim.x + threadIdx.x;
    if (i < n) p[i] = v;
}

// H = 1<<hs (8 or 1); C is 64 or 256
__global__ void att_coef_kernel(const float* __restrict__ xl,
                                const float* __restrict__ a_src,
                                const float* __restrict__ a_dst,
                                float* __restrict__ als, float* __restrict__ ald,
                                int N, int hs, int C)
{
    int gid = blockIdx.x * blockDim.x + threadIdx.x;
    if (gid >= (N << hs)) return;
    int h = gid & ((1 << hs) - 1);
    const float* row = xl + (size_t)gid * C;   // gid*C == (n*H + h)*C
    const float* as  = a_src + h * C;
    const float* ad  = a_dst + h * C;
    float s = 0.f, d = 0.f;
    for (int c = 0; c < C; ++c) { float x = row[c]; s += x * as[c]; d += x * ad[c]; }
    als[gid] = s; ald[gid] = d;
}

// order-independent float atomic max via sign-split int/uint compare
__device__ __forceinline__ void atomicMaxF(float* addr, float val) {
    if (val >= 0.f) atomicMax((int*)addr, __float_as_int(val));
    else            atomicMin((unsigned int*)addr, __float_as_uint(val));
}

__global__ void edge_logit_kernel(const int* __restrict__ src, const int* __restrict__ dst,
                                  const float* __restrict__ als, const float* __restrict__ ald,
                                  float* __restrict__ ebuf, float* __restrict__ mmax,
                                  int E, int N, int hs)
{
    int gid = blockIdx.x * blockDim.x + threadIdx.x;
    if (gid >= ((E + N) << hs)) return;
    int e = gid >> hs, h = gid & ((1 << hs) - 1);
    int s, d;
    if (e < E) { s = src[e]; d = dst[e]; } else { s = d = e - E; }   // self-loops
    float v = als[(s << hs) + h] + ald[(d << hs) + h];
    v = v > 0.f ? v : 0.2f * v;                                     // leaky_relu(0.2)
    ebuf[gid] = v;
    atomicMaxF(&mmax[(d << hs) + h], v);
}

__global__ void edge_exp_kernel(const int* __restrict__ dst, float* __restrict__ ebuf,
                                const float* __restrict__ mmax, float* __restrict__ den,
                                int E, int N, int hs)
{
    int gid = blockIdx.x * blockDim.x + threadIdx.x;
    if (gid >= ((E + N) << hs)) return;
    int e = gid >> hs, h = gid & ((1 << hs) - 1);
    int d = (e < E) ? dst[e] : e - E;
    float ex = __expf(ebuf[gid] - mmax[(d << hs) + h]);
    ebuf[gid] = ex;
    atomicAdd(&den[(d << hs) + h], ex);
}

// one wave per (edge, head): alpha-weighted gather/scatter-add (L2-resident)
__global__ __launch_bounds__(256) void edge_aggr_kernel(
    const int* __restrict__ src, const int* __restrict__ dst,
    const float* __restrict__ xl, const float* __restrict__ ebuf,
    const float* __restrict__ den, float* __restrict__ out,
    int E, int N, int hs, int C)
{
    int w    = blockIdx.x * (blockDim.x >> 5) + (threadIdx.x >> 5);
    int lane = threadIdx.x & 31;
    if (w >= ((E + N) << hs)) return;
    int e = w >> hs, h = w & ((1 << hs) - 1);
    int s, d;
    if (e < E) { s = src[e]; d = dst[e]; } else { s = d = e - E; }
    float alpha = ebuf[w] / den[(d << hs) + h];
    const float* xs = xl  + ((size_t)(s << hs) + h) * C;
    float*       od = out + ((size_t)(d << hs) + h) * C;
    for (int c = lane; c < C; c += 32)
        atomicAdd(&od[c], xs[c] * alpha);
}

__global__ void finalize_kernel(const float* __restrict__ acc, const float* __restrict__ bias,
                                float* __restrict__ dst, int total, int colmask, int relu)
{
    int i = blockIdx.x * blockDim.x + threadIdx.x;
    if (i >= total) return;
    float v = acc[i] + bias[i & colmask];
    if (relu) v = fmaxf(v, 0.f);
    dst[i] = v;
}

// ---------------------------------------------------------------------------
static void run_gat_layer(const float* in, int Fin, const float* W,
                          const float* a_s, const float* a_d, const float* b,
                          int H, int C, int N, int E,
                          const int* src, const int* dst,
                          float* xl, float* outbuf, float* dest,
                          float* als, float* ald, float* mm, float* den, float* eb,
                          int relu, hipStream_t stream)
{
    const int HC = H * C;
    const int hs = (H == 8) ? 3 : 0;
    dim3 gg((HC + BN - 1) / BN, (N + BM - 1) / BM);
    gemm_bf16_wmma<<<gg, 256, 0, stream>>>(in, W, xl, N, HC, Fin, nullptr, 0);

    int NH = N * H;
    fill_kernel<<<(NH + 255) / 256, 256, 0, stream>>>(mm, -INFINITY, NH);
    fill_kernel<<<(NH + 255) / 256, 256, 0, stream>>>(den, 0.f, NH);
    fill_kernel<<<(N * HC + 255) / 256, 256, 0, stream>>>(outbuf, 0.f, N * HC);

    att_coef_kernel<<<(NH + 255) / 256, 256, 0, stream>>>(xl, a_s, a_d, als, ald, N, hs, C);

    int tot = (E + N) * H;
    edge_logit_kernel<<<(tot + 255) / 256, 256, 0, stream>>>(src, dst, als, ald, eb, mm, E, N, hs);
    edge_exp_kernel<<<(tot + 255) / 256, 256, 0, stream>>>(dst, eb, mm, den, E, N, hs);
    edge_aggr_kernel<<<(tot + 7) / 8, 256, 0, stream>>>(src, dst, xl, eb, den, outbuf, E, N, hs, C);

    finalize_kernel<<<(N * HC + 255) / 256, 256, 0, stream>>>(outbuf, b, dest, N * HC, HC - 1, relu);
}

extern "C" void kernel_launch(void* const* d_in, const int* in_sizes, int n_in,
                              void* d_out, int out_size, void* d_ws, size_t ws_size,
                              hipStream_t stream)
{
    const float* x   = (const float*)d_in[0];
    const int*   ei  = (const int*)  d_in[1];
    const float* W0  = (const float*)d_in[2];
    const float* as0 = (const float*)d_in[3];
    const float* ad0 = (const float*)d_in[4];
    const float* b0  = (const float*)d_in[5];
    const float* W1  = (const float*)d_in[6];
    const float* as1 = (const float*)d_in[7];
    const float* ad1 = (const float*)d_in[8];
    const float* b1  = (const float*)d_in[9];
    const float* W2  = (const float*)d_in[10];
    const float* as2 = (const float*)d_in[11];
    const float* ad2 = (const float*)d_in[12];
    const float* b2  = (const float*)d_in[13];
    const float* fcw = (const float*)d_in[14];
    const float* fcb = (const float*)d_in[15];

    const int N = in_sizes[0] / 512;   // 10000
    const int E = in_sizes[1] / 2;     // 320000
    const int* src = ei;
    const int* dst = ei + E;

    float* ws = (float*)d_ws;
    size_t o = 0;
    float* XL  = ws + o; o += (size_t)N * 512;       // xl = x @ W
    float* B1  = ws + o; o += (size_t)N * 512;       // layer activations (ping-pong)
    float* ALS = ws + o; o += (size_t)N * 8;
    float* ALD = ws + o; o += (size_t)N * 8;
    float* MM  = ws + o; o += (size_t)N * 8;
    float* DEN = ws + o; o += (size_t)N * 8;
    float* EB  = ws + o; o += (size_t)(E + N) * 8;   // per-(edge,head) logits/weights

    float* emb    = (float*)d_out;            // [N, 256]
    float* logits = emb + (size_t)N * 256;    // [N, 40]

    // layer 0: x -> B1 (relu)
    run_gat_layer(x,  512, W0, as0, ad0, b0, 8, 64, N, E, src, dst,
                  XL, B1, B1, ALS, ALD, MM, DEN, EB, 1, stream);
    // layer 1: B1 -> B1 (GEMM reads B1 first; accumulator zero-fill comes after, stream-ordered)
    run_gat_layer(B1, 512, W1, as1, ad1, b1, 8, 64, N, E, src, dst,
                  XL, B1, B1, ALS, ALD, MM, DEN, EB, 1, stream);
    // layer 2 (H=1, C=256, mean over 1 head == identity): B1 -> d_out embedding (relu)
    run_gat_layer(B1, 512, W2, as2, ad2, b2, 1, 256, N, E, src, dst,
                  XL, emb, emb, ALS, ALD, MM, DEN, EB, 1, stream);
    // fc: logits = emb @ fcw + fcb
    dim3 gf((40 + BN - 1) / BN, (N + BM - 1) / BM);
    gemm_bf16_wmma<<<gf, 256, 0, stream>>>(emb, fcw, logits, N, 40, 256, fcb, 0);
}
